// Net_91225105367816
// MI455X (gfx1250) — matrix operand order repaired and checked by reading, not compiled, not run on hardware
//
#include <hip/hip_runtime.h>
#include <hip/hip_bf16.h>

typedef __attribute__((ext_vector_type(16))) _Float16 v16h;
typedef __attribute__((ext_vector_type(8)))  float    v8f;

#define SLOPE 0.01f
__device__ __forceinline__ float leaky(float x) { return x >= 0.0f ? x : SLOPE * x; }

// ---------------------------------------------------------------------------
// WMMA fragment loaders (wave32, contiguous k order shared by A and B, f16, 16x16x32)
// A (16x32, MxK): lanes 0-15 -> M=lane, K=0..15; lanes 16-31 -> M=lane-16, K=16..31
// B (32x16, KxN): lanes 0-15 -> N=lane, K=0..15; lanes 16-31 -> N=lane-16, K=16..31
// f16 feature buffers are ld=32, zero padded, so K=32 always.
// ---------------------------------------------------------------------------
__device__ __forceinline__ v16h load_a_frag(const _Float16* __restrict__ base, int lane) {
  const int m  = lane & 15;
  const int kb = (lane >> 4) * 16;
  const _Float16* r = base + (size_t)m * 32;
  v16h a;
#pragma unroll
  for (int i = 0; i < 8; ++i) a[i] = r[kb + i];
#pragma unroll
  for (int i = 0; i < 8; ++i) a[8 + i] = r[kb + 8 + i];
  return a;
}

__device__ __forceinline__ v16h load_b_frag(const _Float16* __restrict__ base, int lane) {
  const int n  = lane & 15;
  const int kb = (lane >> 4) * 16;
  const _Float16* r = base + (size_t)n * 32 + kb;
  v16h b;
#pragma unroll
  for (int i = 0; i < 16; ++i) b[i] = r[i];
  return b;
}

// ---------------------------------------------------------------------------
// Encoder: x_pfc [N,13] -> mlp2 -> enc [N,32] (f32 ld32 + f16 ld32 + norms + charged)
// ---------------------------------------------------------------------------
__global__ __launch_bounds__(256)
void encode_pfc_kernel(const float* __restrict__ x,
                       const float* __restrict__ w1, const float* __restrict__ b1,
                       const float* __restrict__ w2, const float* __restrict__ b2,
                       float* __restrict__ enc, _Float16* __restrict__ ench,
                       float* __restrict__ nrm, int* __restrict__ charged, int n) {
  const int r = blockIdx.x * blockDim.x + threadIdx.x;
  if (r >= n) return;
  float in[13];
#pragma unroll
  for (int t = 0; t < 13; ++t) in[t] = x[(size_t)r * 13 + t];
  float h[32];
#pragma unroll
  for (int j = 0; j < 32; ++j) {
    float s = b1[j];
#pragma unroll
    for (int t = 0; t < 13; ++t) s += in[t] * w1[t * 32 + j];
    h[j] = leaky(s);
  }
  float norm = 0.0f;
#pragma unroll
  for (int j = 0; j < 32; ++j) {
    float s = b2[j];
#pragma unroll
    for (int t = 0; t < 32; ++t) s += h[t] * w2[t * 32 + j];
    enc[(size_t)r * 32 + j]  = s;
    ench[(size_t)r * 32 + j] = (_Float16)s;
    norm += s * s;
  }
  nrm[r] = norm;
  charged[r] = (in[11] != 0.0f) ? 1 : 0;
}

// ---------------------------------------------------------------------------
// Encoder: x_vtx [N,4] -> mlp2 -> directly into d_out region [N,32]
// ---------------------------------------------------------------------------
__global__ __launch_bounds__(256)
void encode_vtx_kernel(const float* __restrict__ x,
                       const float* __restrict__ w1, const float* __restrict__ b1,
                       const float* __restrict__ w2, const float* __restrict__ b2,
                       float* __restrict__ enc, int n) {
  const int r = blockIdx.x * blockDim.x + threadIdx.x;
  if (r >= n) return;
  float in[4];
#pragma unroll
  for (int t = 0; t < 4; ++t) in[t] = x[(size_t)r * 4 + t];
  float h[32];
#pragma unroll
  for (int j = 0; j < 32; ++j) {
    float s = b1[j];
#pragma unroll
    for (int t = 0; t < 4; ++t) s += in[t] * w1[t * 32 + j];
    h[j] = leaky(s);
  }
#pragma unroll
  for (int j = 0; j < 32; ++j) {
    float s = b2[j];
#pragma unroll
    for (int t = 0; t < 32; ++t) s += h[t] * w2[t * 32 + j];
    enc[(size_t)r * 32 + j] = s;
  }
}

// ---------------------------------------------------------------------------
// KNN edge conv: Gram via WMMA -> d2 in LDS (f16) -> iterative top-K=64 ->
// mean_k leaky(base + (x_j - x_i) @ Wj).
// Block = 512 threads (16 waves) handles one 16-row dst tile of one batch.
// ---------------------------------------------------------------------------
template <int D, int O, bool AUX>
__global__ __launch_bounds__(512)
void knn_conv_kernel(const float* __restrict__ xin,      // [B*1024, D] f32
                     const _Float16* __restrict__ xin_h, // [B*1024, 32] f16, zero padded
                     const float* __restrict__ norms,    // [B*1024]
                     const int* __restrict__ valid,      // [B*1024] or nullptr
                     const float* __restrict__ W,        // [2D, O]
                     const float* __restrict__ bias,     // [O]
                     float* __restrict__ xout,           // [B*1024, O]
                     _Float16* __restrict__ xout_h,      // [B*1024, 32] or nullptr
                     float* __restrict__ out_norms) {    // [B*1024] or nullptr
  __shared__ _Float16 s_dist[16][1024];  // 32 KB
  __shared__ int      s_nbr[16][64];     //  4 KB

  const int lane  = threadIdx.x & 31;
  const int wave  = threadIdx.x >> 5;   // 0..15
  const int tile  = blockIdx.x;         // dst 16-row tile, 0..63
  const int batch = blockIdx.y;         // 0..31
  const int dbase = batch * 1024 + tile * 16;
  const int sbase = batch * 1024;

  // ---- Phase 1: Gram tiles via v_wmma_f32_16x16x32_f16, d2 into LDS ----
  const v16h afrag = load_a_frag(xin_h + (size_t)dbase * 32, lane);
  const int mrow = (lane >> 4) * 8;
  const int ncol = lane & 15;
#pragma unroll
  for (int t = 0; t < 4; ++t) {
    const int stile = wave * 4 + t;  // src tile 0..63
    const v16h bfrag = load_b_frag(xin_h + (size_t)(sbase + stile * 16) * 32, lane);
    v8f c = {};
    c = __builtin_amdgcn_wmma_f32_16x16x32_f16(false, afrag, false, bfrag,
                                               (short)0, c, false, false);
    const int col = stile * 16 + ncol;
    const float ns = norms[sbase + col];
    const bool ok = valid ? (valid[sbase + col] != 0) : true;
#pragma unroll
    for (int v = 0; v < 8; ++v) {
      const int m = mrow + v;
      float d2 = norms[dbase + m] - 2.0f * c[v] + ns;
      if (!ok) d2 = 3.0e4f;
      s_dist[m][col] = (_Float16)d2;
    }
  }
  __syncthreads();

  // ---- Phase 2: top-K=64 per dst row (one wave per row) ----
  const int row = wave;
  for (int k = 0; k < 64; ++k) {
    float best = 1.0e30f;
    int bidx = 0x7FFFFFFF;
#pragma unroll
    for (int i = 0; i < 32; ++i) {
      const int cidx = lane + 32 * i;
      const float v = (float)s_dist[row][cidx];
      if (v < best) { best = v; bidx = cidx; }
    }
#pragma unroll
    for (int off = 16; off > 0; off >>= 1) {
      const float ov = __shfl_xor(best, off, 32);
      const int   oi = __shfl_xor(bidx, off, 32);
      if (ov < best || (ov == best && oi < bidx)) { best = ov; bidx = oi; }
    }
    if (lane == 0) s_nbr[row][k] = bidx;
    // the lane that will re-read this slot performs the kill store
    if (lane == (bidx & 31)) s_dist[row][bidx] = (_Float16)3.0e4f;
    __builtin_amdgcn_wave_barrier();
  }
  __syncthreads();

  // ---- Phase 3: message MLP + mean over K (lane = output channel) ----
  const int grow = dbase + row;
  float xir[D];
#pragma unroll
  for (int t = 0; t < D; ++t) xir[t] = xin[(size_t)grow * D + t];

  float outv = 0.0f;
  if (lane < O) {
    float base = bias[lane];
#pragma unroll
    for (int t = 0; t < D; ++t) base += xir[t] * W[t * O + lane];
    float acc = 0.0f;
    for (int k = 0; k < 64; ++k) {
      const int j = s_nbr[row][k];
      const float* xj = xin + (size_t)(sbase + j) * D;
      if (k + 1 < 64) {
        const int jn = s_nbr[row][k + 1];
        __builtin_prefetch(xin + (size_t)(sbase + jn) * D, 0, 1);
      }
      float s = base;
#pragma unroll
      for (int t = 0; t < D; ++t) s += (xj[t] - xir[t]) * W[(D + t) * O + lane];
      acc += leaky(s);
    }
    outv = acc * (1.0f / 64.0f);
    xout[(size_t)grow * O + lane] = outv;
  }
  if (AUX) {
    xout_h[(size_t)grow * 32 + lane] = (_Float16)((lane < O) ? outv : 0.0f);
    float sq = (lane < O) ? outv * outv : 0.0f;
#pragma unroll
    for (int off = 16; off > 0; off >>= 1) sq += __shfl_xor(sq, off, 32);
    if (lane == 0) out_norms[grow] = sq;
  }
}

// ---------------------------------------------------------------------------
// Output head: 32 -> 64 -> 32 -> 4 -> 1, weights staged in LDS.
// Also writes batch ids (row/1024) as floats.
// ---------------------------------------------------------------------------
__global__ __launch_bounds__(256)
void head_kernel(const float* __restrict__ f2,  // [N,32]
                 const float* __restrict__ w1, const float* __restrict__ b1,  // [32,64],[64]
                 const float* __restrict__ w2, const float* __restrict__ b2,  // [64,32],[32]
                 const float* __restrict__ w3, const float* __restrict__ b3,  // [32,4],[4]
                 const float* __restrict__ w4, const float* __restrict__ b4,  // [4,1],[1]
                 float* __restrict__ out, float* __restrict__ outb, int n) {
  __shared__ float sw1[32 * 64];
  __shared__ float sw2[64 * 32];
  __shared__ float sw3[32 * 4];
  for (int i = threadIdx.x; i < 32 * 64; i += 256) sw1[i] = w1[i];
  for (int i = threadIdx.x; i < 64 * 32; i += 256) sw2[i] = w2[i];
  for (int i = threadIdx.x; i < 32 * 4;  i += 256) sw3[i] = w3[i];
  __syncthreads();

  const int r = blockIdx.x * blockDim.x + threadIdx.x;
  if (r >= n) return;
  float x[32];
#pragma unroll
  for (int t = 0; t < 32; ++t) x[t] = f2[(size_t)r * 32 + t];
  float h1[64];
#pragma unroll
  for (int j = 0; j < 64; ++j) {
    float s = b1[j];
#pragma unroll
    for (int t = 0; t < 32; ++t) s += x[t] * sw1[t * 64 + j];
    h1[j] = leaky(s);
  }
  float h2[32];
#pragma unroll
  for (int j = 0; j < 32; ++j) {
    float s = b2[j];
#pragma unroll
    for (int t = 0; t < 64; ++t) s += h1[t] * sw2[t * 32 + j];
    h2[j] = leaky(s);
  }
  float h3[4];
#pragma unroll
  for (int j = 0; j < 4; ++j) {
    float s = b3[j];
#pragma unroll
    for (int t = 0; t < 32; ++t) s += h2[t] * sw3[t * 4 + j];
    h3[j] = leaky(s);
  }
  float o = b4[0];
#pragma unroll
  for (int t = 0; t < 4; ++t) o += h3[t] * w4[t];
  out[r]  = o;
  outb[r] = (float)(r >> 10);  // batch id (NPF = 1024)
}

// ---------------------------------------------------------------------------
// Launch
// ---------------------------------------------------------------------------
extern "C" void kernel_launch(void* const* d_in, const int* in_sizes, int n_in,
                              void* d_out, int out_size, void* d_ws, size_t ws_size,
                              hipStream_t stream) {
  (void)in_sizes; (void)n_in; (void)out_size; (void)ws_size;
  const float* x_pfc   = (const float*)d_in[0];
  const float* x_vtx   = (const float*)d_in[1];
  const float* pfc_w1  = (const float*)d_in[4];
  const float* pfc_b1  = (const float*)d_in[5];
  const float* pfc_w2  = (const float*)d_in[6];
  const float* pfc_b2  = (const float*)d_in[7];
  const float* vtx_w1  = (const float*)d_in[8];
  const float* vtx_b1  = (const float*)d_in[9];
  const float* vtx_w2  = (const float*)d_in[10];
  const float* vtx_b2  = (const float*)d_in[11];
  const float* conv1_w = (const float*)d_in[12];
  const float* conv1_b = (const float*)d_in[13];
  const float* conv2_w = (const float*)d_in[14];
  const float* conv2_b = (const float*)d_in[15];
  const float* out_w1  = (const float*)d_in[16];
  const float* out_b1  = (const float*)d_in[17];
  const float* out_w2  = (const float*)d_in[18];
  const float* out_b2  = (const float*)d_in[19];
  const float* out_w3  = (const float*)d_in[20];
  const float* out_b3  = (const float*)d_in[21];
  const float* out_w4  = (const float*)d_in[22];
  const float* out_b4  = (const float*)d_in[23];

  const int NPFC = 32 * 1024;  // 32768 pfc rows
  const int NV   = 32 * 32;    // 1024 vtx rows

  float* out      = (float*)d_out;
  float* o_pred   = out;                         // [32768]
  float* o_batch  = out + NPFC;                  // [32768]
  float* o_feats1 = out + 2 * NPFC;              // [32768,16]
  float* o_vtx    = out + 2 * NPFC + NPFC * 16;  // [1024,32]

  char* ws = (char*)d_ws;
  float*    encF    = (float*)ws;     ws += (size_t)NPFC * 32 * sizeof(float);
  _Float16* encH    = (_Float16*)ws;  ws += (size_t)NPFC * 32 * sizeof(_Float16);
  float*    nrm1    = (float*)ws;     ws += (size_t)NPFC * sizeof(float);
  int*      charged = (int*)ws;       ws += (size_t)NPFC * sizeof(int);
  _Float16* f1H     = (_Float16*)ws;  ws += (size_t)NPFC * 32 * sizeof(_Float16);
  float*    nrm2    = (float*)ws;     ws += (size_t)NPFC * sizeof(float);
  float*    f2F     = (float*)ws;     ws += (size_t)NPFC * 32 * sizeof(float);

  encode_pfc_kernel<<<NPFC / 256, 256, 0, stream>>>(
      x_pfc, pfc_w1, pfc_b1, pfc_w2, pfc_b2, encF, encH, nrm1, charged, NPFC);

  encode_vtx_kernel<<<NV / 256, 256, 0, stream>>>(
      x_vtx, vtx_w1, vtx_b1, vtx_w2, vtx_b2, o_vtx, NV);

  // conv1: D=32, O=16; writes feats1 straight into d_out + aux (f16 copy + norms)
  knn_conv_kernel<32, 16, true><<<dim3(64, 32), 512, 0, stream>>>(
      encF, encH, nrm1, nullptr, conv1_w, conv1_b, o_feats1, f1H, nrm2);

  // conv2: D=16, O=32; src masked by charged
  knn_conv_kernel<16, 32, false><<<dim3(64, 32), 512, 0, stream>>>(
      o_feats1, f1H, nrm2, charged, conv2_w, conv2_b, f2F, nullptr, nullptr);

  head_kernel<<<NPFC / 256, 256, 0, stream>>>(
      f2F, out_w1, out_b1, out_w2, out_b2, out_w3, out_b3, out_w4, out_b4,
      o_pred, o_batch, NPFC);
}
